// DecoderBlock_44590350467483
// MI455X (gfx1250) — matrix operand order repaired and checked by reading, not compiled, not run on hardware
//
#include <hip/hip_runtime.h>
#include <hip/hip_bf16.h>

typedef __attribute__((ext_vector_type(16))) _Float16 v16h;
typedef __attribute__((ext_vector_type(8)))  _Float16 v8h;
typedef __attribute__((ext_vector_type(8)))  float    v8f;
typedef __attribute__((ext_vector_type(4)))  int      v4i;

#define D_MODEL 1024
#define NHEAD   16
#define DK      64
#define BATCH   4
#define LTOK    1024
#define NROWS   4096   /* LT*B == LS*B */
#define FFN_D   4096
#define NEGBIG  (-10000000000.0f)

#define WMMA_F16(a, b, c) \
  __builtin_amdgcn_wmma_f32_16x16x32_f16(false, (a), false, (b), (short)0, (c), false, false)

union V16u { v16h v; v8h h[2]; };

// ---- global -> LDS movement: use gfx1250 async-to-LDS path if the builtin
// exists on this toolchain, else fall back to load+store (still correct).
#if __has_builtin(__builtin_amdgcn_global_load_async_to_lds_b128)
#define HAVE_ASYNC_LDS 1
typedef __attribute__((address_space(1))) v4i* gas_v4i;
typedef __attribute__((address_space(3))) v4i* las_v4i;
#else
#define HAVE_ASYNC_LDS 0
#endif

__device__ inline void copy16_g2l(const _Float16* g, _Float16* l) {
#if HAVE_ASYNC_LDS
  __builtin_amdgcn_global_load_async_to_lds_b128(
      (gas_v4i)(void*)g, (las_v4i)(void*)l, 0, 0);
#else
  *(v8h*)l = *(const v8h*)g;
#endif
}

__device__ inline void g2l_join() {
#if HAVE_ASYNC_LDS
#if __has_builtin(__builtin_amdgcn_s_wait_asynccnt)
  __builtin_amdgcn_s_wait_asynccnt(0);
#else
  asm volatile("s_wait_asynccnt 0x0" ::: "memory");
#endif
#endif
}

// ---- WMMA operand loaders (CDNA5 16-bit layouts, wave32) ----------------
// A 16x32 f16 tile: lane = M row (mod 16); lanes<16 hold K {0..7,16..23},
// lanes>=16 hold K {8..15,24..31}, 2 halves per VGPR in order.
__device__ inline v16h ldA(const _Float16* __restrict__ base, int ldRow, int kbase) {
  int lane = threadIdx.x & 31;
  int row  = lane & 15;
  int k0   = kbase + ((lane < 16) ? 0 : 8);
  const _Float16* p = base + (size_t)row * ldRow + k0;
  V16u u;
  u.h[0] = *(const v8h*)(p);
  u.h[1] = *(const v8h*)(p + 16);
  return u.v;
}

// B 32x16 f16 tile where B[k][n] = base[n*ldRow + k] (i.e. row-major W for X*W^T):
// lane = N col (mod 16); lanes<16 hold K {0..7,16..23}, lanes>=16 hold K {8..15,24..31}.
__device__ inline v16h ldB(const _Float16* __restrict__ base, int ldRow, int kbase) {
  int lane = threadIdx.x & 31;
  int n    = lane & 15;
  int k0   = kbase + ((lane < 16) ? 0 : 8);
  const _Float16* p = base + (size_t)n * ldRow + k0;
  V16u u;
  u.h[0] = *(const v8h*)(p);
  u.h[1] = *(const v8h*)(p + 16);
  return u.v;
}

// B operand for O += P*V from LDS V tile (key-major): gather down a column.
__device__ inline v16h ldBV(const _Float16 (*Vl)[64], int dtile) {
  int lane = threadIdx.x & 31;
  int n    = dtile * 16 + (lane & 15);
  int kb   = (lane < 16) ? 0 : 8;
  v16h r;
#pragma unroll
  for (int j = 0; j < 16; ++j) r[j] = Vl[kb + j + ((j < 8) ? 0 : 8)][n];
  return r;
}

// ---- elementwise cast ---------------------------------------------------
__global__ void cast_f32_to_f16(const float* __restrict__ x, _Float16* __restrict__ y, int n) {
  int i = blockIdx.x * blockDim.x + threadIdx.x;
  if (i < n) y[i] = (_Float16)x[i];
}

// ---- LayerNorm (per 1024-wide row) -> f16 -------------------------------
__global__ void __launch_bounds__(256) layernorm_to_f16(
    const float* __restrict__ x, const float* __restrict__ g,
    const float* __restrict__ bt, _Float16* __restrict__ y) {
  __shared__ float s1[256], s2[256];
  int row = blockIdx.x;
  const float* xr = x + (size_t)row * D_MODEL;
  float a = 0.0f, b = 0.0f;
  for (int i = threadIdx.x; i < D_MODEL; i += 256) { float v = xr[i]; a += v; b += v * v; }
  s1[threadIdx.x] = a; s2[threadIdx.x] = b;
  __syncthreads();
  for (int off = 128; off > 0; off >>= 1) {
    if ((int)threadIdx.x < off) { s1[threadIdx.x] += s1[threadIdx.x + off]; s2[threadIdx.x] += s2[threadIdx.x + off]; }
    __syncthreads();
  }
  float mean = s1[0] * (1.0f / D_MODEL);
  float var  = s2[0] * (1.0f / D_MODEL) - mean * mean;
  float inv  = rsqrtf(var + 1e-5f);
  for (int i = threadIdx.x; i < D_MODEL; i += 256)
    y[(size_t)row * D_MODEL + i] = (_Float16)((xr[i] - mean) * inv * g[i] + bt[i]);
}

// ---- WMMA GEMM:  Y = X (NxK, f16) * W^T (W is MxK f16) + bias  ----------
// 8 waves/block share one 64-wide column stripe; the 64x32 W chunk is staged
// in LDS once per K-step (async-to-LDS when available) and shared by all 8
// waves, cutting the dominant operand's global traffic 8x.
// Requires N % 128 == 0 (uniform barriers; no early exit).
template <bool RELU>
__global__ void __launch_bounds__(256) gemm_wmma(
    const _Float16* __restrict__ A, const _Float16* __restrict__ W,
    const float* __restrict__ bias, const float* __restrict__ resid,
    float* __restrict__ outF, _Float16* __restrict__ outH,
    int N, int K, int M) {
  __shared__ __align__(16) _Float16 Wl[64][32];
  int tid  = threadIdx.x;
  int wave = tid >> 5;
  int lane = tid & 31;
  int row0 = (blockIdx.y * 8 + wave) * 16;
  int col0 = blockIdx.x * 64;

  v8f c0 = {}, c1 = {}, c2 = {}, c3 = {};
  const _Float16* Abase = A + (size_t)row0 * K;
  // staging assignment: thread t copies 8 halves of W row (col0 + t/4)
  int tr = tid >> 2;            // 0..63
  int tp = (tid & 3) * 8;       // 0,8,16,24
  const _Float16* Wg = W + (size_t)(col0 + tr) * K + tp;
  _Float16* Wdst = &Wl[tr][tp];

  for (int k = 0; k < K; k += 32) {
    __syncthreads();                 // previous chunk fully consumed
    copy16_g2l(Wg + k, Wdst);
    g2l_join();
    __syncthreads();                 // W chunk visible to all waves
    v16h a  = ldA(Abase, K, k);
    v16h b0 = ldB(&Wl[0][0],  32, 0);
    v16h b1 = ldB(&Wl[16][0], 32, 0);
    v16h b2 = ldB(&Wl[32][0], 32, 0);
    v16h b3 = ldB(&Wl[48][0], 32, 0);
    c0 = WMMA_F16(a, b0, c0);
    c1 = WMMA_F16(a, b1, c1);
    c2 = WMMA_F16(a, b2, c2);
    c3 = WMMA_F16(a, b3, c3);
  }
  int rowOff = (lane < 16) ? 0 : 8;
  int nl = lane & 15;
#pragma unroll
  for (int t = 0; t < 4; ++t) {
    int col = col0 + t * 16 + nl;
    float bv = bias ? bias[col] : 0.0f;
    v8f cc = (t == 0) ? c0 : (t == 1) ? c1 : (t == 2) ? c2 : c3;
#pragma unroll
    for (int j = 0; j < 8; ++j) {
      int row = row0 + rowOff + j;
      float v = cc[j] + bv;
      if (RELU) v = fmaxf(v, 0.0f);
      if (resid) v += resid[(size_t)row * M + col];
      if (outF) outF[(size_t)row * M + col] = v;
      if (outH) outH[(size_t)row * M + col] = (_Float16)v;
    }
  }
}

// ---- Flash attention: one block per (b,h, 64-query chunk); 4 waves x 16-query tiles -----
__global__ void __launch_bounds__(128) attention_flash(
    const _Float16* __restrict__ Q, const _Float16* __restrict__ Kv,
    const _Float16* __restrict__ Vv, const unsigned char* __restrict__ smask,
    _Float16* __restrict__ outH) {
  __shared__ __align__(16) _Float16 Kl[32][64];
  __shared__ __align__(16) _Float16 Vl[32][64];
  __shared__ __align__(16) _Float16 Pl[4][16][32];

  int bh = blockIdx.x;
  int b  = bh >> 4;   // batch
  int h  = bh & 15;   // head
  int wave = threadIdx.x >> 5;
  int lane = threadIdx.x & 31;
  int q0 = (blockIdx.y * 4 + wave) * 16;

  const size_t ldrow = (size_t)BATCH * D_MODEL;
  const _Float16* Qb = Q + ((size_t)q0 * BATCH + b) * D_MODEL + h * DK;

  // Preload Q tile (16x64) in A layout for the two 32-wide K chunks.
  v16h aq0, aq1;
  {
    int row = lane & 15;
    int k0  = (lane < 16) ? 0 : 8;
    const _Float16* p = Qb + (size_t)row * ldrow;
    V16u u0, u1;
    u0.h[0] = *(const v8h*)(p + k0);
    u0.h[1] = *(const v8h*)(p + k0 + 16);
    u1.h[0] = *(const v8h*)(p + 32 + k0);
    u1.h[1] = *(const v8h*)(p + 32 + k0 + 16);
    aq0 = u0.v; aq1 = u1.v;
  }

  float m[8], l[8];
  v8f o0 = {}, o1 = {}, o2 = {}, o3 = {};
#pragma unroll
  for (int j = 0; j < 8; ++j) { m[j] = -3.0e38f; l[j] = 0.0f; }

  for (int kc = 0; kc < LTOK; kc += 32) {
    __syncthreads();  // previous chunk's LDS consumers are done
    {
      int key  = threadIdx.x >> 2;         // 0..31
      int part = (threadIdx.x & 3) * 16;   // 0,16,32,48
      const _Float16* kp = Kv + ((size_t)(kc + key) * BATCH + b) * D_MODEL + h * DK + part;
      const _Float16* vp = Vv + ((size_t)(kc + key) * BATCH + b) * D_MODEL + h * DK + part;
      copy16_g2l(kp,     &Kl[key][part]);
      copy16_g2l(kp + 8, &Kl[key][part + 8]);
      copy16_g2l(vp,     &Vl[key][part]);
      copy16_g2l(vp + 8, &Vl[key][part + 8]);
    }
    g2l_join();
    __syncthreads();

    // S = Q*K^T for 32 keys (two 16-key N tiles, dk=64 -> two K=32 chunks)
    v8f s0 = {}, s1 = {};
    {
      v16h b00 = ldB(&Kl[0][0],  64, 0);
      v16h b01 = ldB(&Kl[0][0],  64, 32);
      v16h b10 = ldB(&Kl[16][0], 64, 0);
      v16h b11 = ldB(&Kl[16][0], 64, 32);
      s0 = WMMA_F16(aq0, b00, s0);
      s0 = WMMA_F16(aq1, b01, s0);
      s1 = WMMA_F16(aq0, b10, s1);
      s1 = WMMA_F16(aq1, b11, s1);
    }

    // scale + mask + online softmax (row reductions over 16-lane halves)
    bool msk0 = smask[(size_t)(kc + (lane & 15)) * BATCH + b] != 0;
    bool msk1 = smask[(size_t)(kc + 16 + (lane & 15)) * BATCH + b] != 0;
    float p0[8], p1[8], alpha[8];
#pragma unroll
    for (int j = 0; j < 8; ++j) {
      float x0 = msk0 ? NEGBIG : s0[j] * 0.125f;   // 1/sqrt(64)
      float x1 = msk1 ? NEGBIG : s1[j] * 0.125f;
      float mx = fmaxf(x0, x1);
      mx = fmaxf(mx, __shfl_xor(mx, 1, 32));
      mx = fmaxf(mx, __shfl_xor(mx, 2, 32));
      mx = fmaxf(mx, __shfl_xor(mx, 4, 32));
      mx = fmaxf(mx, __shfl_xor(mx, 8, 32));
      float mn = fmaxf(m[j], mx);
      alpha[j] = __expf(m[j] - mn);
      m[j] = mn;
      p0[j] = __expf(x0 - mn);
      p1[j] = __expf(x1 - mn);
      float rs = p0[j] + p1[j];
      rs += __shfl_xor(rs, 1, 32);
      rs += __shfl_xor(rs, 2, 32);
      rs += __shfl_xor(rs, 4, 32);
      rs += __shfl_xor(rs, 8, 32);
      l[j] = l[j] * alpha[j] + rs;
    }
#pragma unroll
    for (int j = 0; j < 8; ++j) {
      o0[j] = o0[j] * alpha[j];
      o1[j] = o1[j] * alpha[j];
      o2[j] = o2[j] * alpha[j];
      o3[j] = o3[j] * alpha[j];
    }

    // C-layout -> A-layout transpose of P (16x32) through per-wave LDS tile
    {
      int rr = (lane < 16) ? 0 : 8;
      int cc = lane & 15;
#pragma unroll
      for (int j = 0; j < 8; ++j) {
        Pl[wave][rr + j][cc]      = (_Float16)p0[j];
        Pl[wave][rr + j][cc + 16] = (_Float16)p1[j];
      }
    }
    __syncthreads();  // conservative: make P stores visible before A-layout reads

    v16h ap = ldA(&Pl[wave][0][0], 32, 0);
    o0 = WMMA_F16(ap, ldBV(Vl, 0), o0);
    o1 = WMMA_F16(ap, ldBV(Vl, 1), o1);
    o2 = WMMA_F16(ap, ldBV(Vl, 2), o2);
    o3 = WMMA_F16(ap, ldBV(Vl, 3), o3);
  }

  // normalize and store to attn buffer (rows = q*B+b, cols = h*64+d), f16
  int rr = (lane < 16) ? 0 : 8;
  int nl = lane & 15;
#pragma unroll
  for (int j = 0; j < 8; ++j) {
    float invl = 1.0f / l[j];
    int q = q0 + rr + j;
    size_t ro = ((size_t)q * BATCH + b) * D_MODEL + h * DK + nl;
    outH[ro]      = (_Float16)(o0[j] * invl);
    outH[ro + 16] = (_Float16)(o1[j] * invl);
    outH[ro + 32] = (_Float16)(o2[j] * invl);
    outH[ro + 48] = (_Float16)(o3[j] * invl);
  }
}

// -------------------------------------------------------------------------
extern "C" void kernel_launch(void* const* d_in, const int* in_sizes, int n_in,
                              void* d_out, int out_size, void* d_ws, size_t ws_size,
                              hipStream_t stream) {
  (void)in_sizes; (void)n_in; (void)out_size; (void)ws_size;
  const float* enc  = (const float*)d_in[0];
  const float* emb  = (const float*)d_in[1];
  const unsigned char* smask = (const unsigned char*)d_in[2];
  // sublayer 0 (mmha, d_in[4..11], ln0) is dead code in the reference -- skipped.
  const float* wq = (const float*)d_in[12]; const float* bq = (const float*)d_in[13];
  const float* wk = (const float*)d_in[14]; const float* bk = (const float*)d_in[15];
  const float* wv = (const float*)d_in[16]; const float* bv = (const float*)d_in[17];
  const float* wo = (const float*)d_in[18]; const float* bo = (const float*)d_in[19];
  const float* w1 = (const float*)d_in[20]; const float* b1 = (const float*)d_in[21];
  const float* w2 = (const float*)d_in[22]; const float* b2 = (const float*)d_in[23];
  const float* ln1g = (const float*)d_in[26]; const float* ln1b = (const float*)d_in[27];
  const float* ln2g = (const float*)d_in[28]; const float* ln2b = (const float*)d_in[29];

  // ---- workspace layout (~88 MB; FFN hidden overlays dead enc/K/V/attn) --
  char* ws = (char*)d_ws;
  size_t off = 0;
  auto alloc = [&](size_t bytes) { char* p = ws + off; off = (off + bytes + 255) & ~(size_t)255; return p; };
  const size_t ACT_H = (size_t)NROWS * D_MODEL * sizeof(_Float16);  // 8 MB, multiple of 256
  _Float16* encH  = (_Float16*)alloc(ACT_H);   // dead after V GEMM; part of hH overlay
  _Float16* Kh    = (_Float16*)alloc(ACT_H);   // dead after attn;   part of hH overlay
  _Float16* Vh    = (_Float16*)alloc(ACT_H);   // dead after attn;   part of hH overlay
  _Float16* attnH = (_Float16*)alloc(ACT_H);   // dead after proj;   part of hH overlay
  _Float16* qinH  = (_Float16*)alloc(ACT_H);   // dead after Q GEMM; reused as zH
  _Float16* Qh    = (_Float16*)alloc(ACT_H);
  float*    encdec= (float*)   alloc((size_t)NROWS * D_MODEL * sizeof(float));
  _Float16* wqH   = (_Float16*)alloc((size_t)D_MODEL * D_MODEL * 2);
  _Float16* wkH   = (_Float16*)alloc((size_t)D_MODEL * D_MODEL * 2);
  _Float16* wvH   = (_Float16*)alloc((size_t)D_MODEL * D_MODEL * 2);
  _Float16* woH   = (_Float16*)alloc((size_t)D_MODEL * D_MODEL * 2);
  _Float16* w1H   = (_Float16*)alloc((size_t)FFN_D * D_MODEL * 2);
  _Float16* w2H   = (_Float16*)alloc((size_t)D_MODEL * FFN_D * 2);
  _Float16* hH = encH;   // 4 x 8MB contiguous = 4096x4096 f16
  _Float16* zH = qinH;

  const int nDD = D_MODEL * D_MODEL;      // 1M
  const int nFD = FFN_D * D_MODEL;        // 4M
  const int nAct = NROWS * D_MODEL;       // 4M

  // ---- stage 0: casts -----------------------------------------------------
  cast_f32_to_f16<<<(nAct + 255) / 256, 256, 0, stream>>>(enc, encH, nAct);
  cast_f32_to_f16<<<(nDD + 255) / 256, 256, 0, stream>>>(wq, wqH, nDD);
  cast_f32_to_f16<<<(nDD + 255) / 256, 256, 0, stream>>>(wk, wkH, nDD);
  cast_f32_to_f16<<<(nDD + 255) / 256, 256, 0, stream>>>(wv, wvH, nDD);
  cast_f32_to_f16<<<(nDD + 255) / 256, 256, 0, stream>>>(wo, woH, nDD);
  cast_f32_to_f16<<<(nFD + 255) / 256, 256, 0, stream>>>(w1, w1H, nFD);
  cast_f32_to_f16<<<(nFD + 255) / 256, 256, 0, stream>>>(w2, w2H, nFD);

  // ---- stage 1: q_in = LN1(embedded) --------------------------------------
  layernorm_to_f16<<<NROWS, 256, 0, stream>>>(emb, ln1g, ln1b, qinH);

  // ---- stage 2: Q/K/V projections (WMMA) ----------------------------------
  dim3 gD(D_MODEL / 64, NROWS / 128);     // 16 x 32
  gemm_wmma<false><<<gD, 256, 0, stream>>>(qinH, wqH, bq, nullptr, nullptr, Qh, NROWS, D_MODEL, D_MODEL);
  gemm_wmma<false><<<gD, 256, 0, stream>>>(encH, wkH, bk, nullptr, nullptr, Kh, NROWS, D_MODEL, D_MODEL);
  gemm_wmma<false><<<gD, 256, 0, stream>>>(encH, wvH, bv, nullptr, nullptr, Vh, NROWS, D_MODEL, D_MODEL);

  // ---- stage 3: attention -------------------------------------------------
  attention_flash<<<dim3(BATCH * NHEAD, LTOK / 64), 128, 0, stream>>>(Qh, Kh, Vh, smask, attnH);

  // ---- stage 4: out-proj + residual(embedded) -> enc_dec (f32) ------------
  gemm_wmma<false><<<gD, 256, 0, stream>>>(attnH, woH, bo, emb, encdec, nullptr, NROWS, D_MODEL, D_MODEL);

  // ---- stage 5: LN2 -> z --------------------------------------------------
  layernorm_to_f16<<<NROWS, 256, 0, stream>>>(encdec, ln2g, ln2b, zH);

  // ---- stage 6: FFN -------------------------------------------------------
  dim3 gF1(FFN_D / 64, NROWS / 128);      // 64 x 32
  gemm_wmma<true><<<gF1, 256, 0, stream>>>(zH, w1H, b1, nullptr, nullptr, hH, NROWS, D_MODEL, FFN_D);
  gemm_wmma<false><<<gD, 256, 0, stream>>>(hH, w2H, b2, encdec, (float*)d_out, nullptr, NROWS, FFN_D, D_MODEL);
}
